// AttentionBlock_57715770524076
// MI455X (gfx1250) — matrix operand order repaired and checked by reading, not compiled, not run on hardware
//
#include <hip/hip_runtime.h>
#include <hip/hip_bf16.h>
#include <math.h>

// ---------------------------------------------------------------------------
// Problem constants (from the reference)
// ---------------------------------------------------------------------------
#define B_      2
#define L_      4096
#define DM_     1024
#define H_      16
#define DK_     64
#define R_      4
#define C_      64          // bucket / chunk size
#define NB_     64          // number of buckets == L/C
#define NCH_    64          // number of chunks  == L/C

typedef _Float16     v8h  __attribute__((ext_vector_type(8)));
typedef _Float16     v16h __attribute__((ext_vector_type(16)));
typedef float        v8f  __attribute__((ext_vector_type(8)));
typedef unsigned int u32;
typedef u32          v4u  __attribute__((ext_vector_type(4)));
typedef u32          v8u  __attribute__((ext_vector_type(8)));

// ---------------------------------------------------------------------------
// WMMA fragment helpers (gfx1250 wave32 layouts, CDNA5 ISA §7.12.2)
//   A (16x32 f16): lane l -> M = l&15 ; element e -> K = e + 8*(e>>3) + 8*(l>>4)
//   B (32x16 f16): lane l -> N = l&15 ; element e -> K = e + 8*(e>>3) + 8*(l>>4)
//     (LDS holds B K-contiguous, i.e. stored transposed as [n][k])
//   C/D (16x16 f32, 8 VGPRs): vgpr r, lane l -> M = r + 8*(l>>4), N = l&15.
// ---------------------------------------------------------------------------
__device__ __forceinline__ v16h cat8(v8h lo, v8h hi) {
    return __builtin_shufflevector(lo, hi, 0,1,2,3,4,5,6,7,8,9,10,11,12,13,14,15);
}

__device__ __forceinline__ v16h load_a_frag(const _Float16* base, int stride,
                                            int mBase, int k0) {
    const int lane = threadIdx.x & 31;
    const int m    = mBase + (lane & 15);
    const int k    = k0 + 8 * (lane >> 4);
    const _Float16* p = base + (size_t)m * stride + k;
    v8h lo = *(const v8h*)(p);
    v8h hi = *(const v8h*)(p + 16);
    return cat8(lo, hi);
}

__device__ __forceinline__ v16h load_b_frag(const _Float16* base, int stride,
                                            int nBase, int k0) {
    const int lane = threadIdx.x & 31;
    const int n    = nBase + (lane & 15);
    const int k    = k0 + 8 * (lane >> 4);
    const _Float16* p = base + (size_t)n * stride + k;
    return cat8(*(const v8h*)(p), *(const v8h*)(p + 16));
}

__device__ __forceinline__ v8f wmma_f16(v16h a, v16h b, v8f c) {
    return __builtin_amdgcn_wmma_f32_16x16x32_f16(false, a, false, b,
                                                  (short)0, c, false, false);
}

// ---------------------------------------------------------------------------
// Tensor Data Mover: 2D f16 tile (tile_w x tile_h elements) from global
// (row stride = row_stride elements) into LDS at byte offset lds_off,
// with optional LDS row padding (CDNA5 ISA).
// Descriptor groups passed in SGPR tuples; groups 2/3 zeroed (2D tensor:
// tile_dim3/tile_dim4 = 0 -> unused). Issue from ONE wave only; completion
// via TENSORcnt.
// ---------------------------------------------------------------------------
__device__ __forceinline__ void tdm_load_tile_f16(const _Float16* gsrc,
                                                  u32 lds_off,
                                                  u32 tile_w, u32 tile_h,
                                                  u32 tensor_w, u32 tensor_h,
                                                  u32 row_stride,
                                                  u32 pad_en,
                                                  u32 pad_interval_code,
                                                  u32 pad_amount_code) {
    const unsigned long long ga = (unsigned long long)(uintptr_t)gsrc;
    v4u g0;
    g0.x = 0x1u;                                   // count=1, user mode
    g0.y = lds_off;                                // LDS byte address
    g0.z = (u32)ga;                                // global_addr[31:0]
    g0.w = (u32)((ga >> 32) & 0x1FFFFFFull)        // global_addr[56:32]
         | (2u << 30);                             // type = 2 ("image")
    v8u g1;
    g1[0] = (1u << 16)                             // data_size = 1 (2 bytes)
          | (pad_en << 20)
          | (pad_interval_code << 22)
          | (pad_amount_code << 25);
    g1[1] = (tensor_w & 0xFFFFu) << 16;            // tensor_dim0[15:0]
    g1[2] = (tensor_w >> 16) | ((tensor_h & 0xFFFFu) << 16);
    g1[3] = (tensor_h >> 16) | (tile_w << 16);     // tile_dim0
    g1[4] = tile_h;                                // tile_dim1 (tile_dim2 = 0)
    g1[5] = row_stride;                            // tensor_dim0_stride[31:0]
    g1[6] = 0;
    g1[7] = 0;
    const v4u gz = {0u, 0u, 0u, 0u};
    asm volatile("tensor_load_to_lds %0, %1, %2, %3"
                 :: "s"(g0), "s"(g1), "s"(gz), "s"(gz)
                 : "memory");
}

// ---------------------------------------------------------------------------
// Kernel 1: f32 -> f16 conversion (weights / activations)
// ---------------------------------------------------------------------------
__global__ void cvt_f32_f16_kernel(const float* __restrict__ src,
                                   _Float16* __restrict__ dst, int n) {
    int i = blockIdx.x * blockDim.x + threadIdx.x;
    if (i < n) dst[i] = (_Float16)src[i];
}

// ---------------------------------------------------------------------------
// Kernel 2: LayerNorm over D_MODEL, f32 in -> f16 out. One block per row.
// ---------------------------------------------------------------------------
__global__ __launch_bounds__(256)
void layernorm_kernel(const float* __restrict__ x,
                      const float* __restrict__ gamma,
                      const float* __restrict__ beta,
                      _Float16* __restrict__ out) {
    const int row = blockIdx.x;                 // B*L rows
    const float* xr = x + (size_t)row * DM_;
    __shared__ float red[256];
    float s = 0.f, s2 = 0.f;
    for (int i = threadIdx.x; i < DM_; i += 256) {
        float v = xr[i]; s += v; s2 += v * v;
    }
    red[threadIdx.x] = s; __syncthreads();
    for (int o = 128; o > 0; o >>= 1) {
        if (threadIdx.x < o) red[threadIdx.x] += red[threadIdx.x + o];
        __syncthreads();
    }
    const float mean = red[0] * (1.0f / DM_);
    __syncthreads();
    red[threadIdx.x] = s2; __syncthreads();
    for (int o = 128; o > 0; o >>= 1) {
        if (threadIdx.x < o) red[threadIdx.x] += red[threadIdx.x + o];
        __syncthreads();
    }
    const float var = red[0] * (1.0f / DM_) - mean * mean;
    const float inv = rsqrtf(var + 1e-5f);
    _Float16* orow = out + (size_t)row * DM_;
    for (int i = threadIdx.x; i < DM_; i += 256)
        orow[i] = (_Float16)((xr[i] - mean) * inv * gamma[i] + beta[i]);
}

// ---------------------------------------------------------------------------
// Kernel 3: WMMA GEMM  C[M,N] = A[M,K](f16) * W[K,N](f16) + bias, f32 out.
//   Block tile 64x64, 128 threads (4 waves), K stepped by 32.
//   A tile staged by the Tensor Data Mover (async DMA, TENSORcnt),
//   with TDM LDS-padding reproducing the 48-half padded rows:
//     row = 32 halves = 16 DWORDs -> pad_interval code 3 (16 DWORDs),
//     pad 8 halves = 4 DWORDs     -> pad_amount  code 3 (4 DWORDs).
//   mode 0: out row-major [M,N]
//   mode 1: scatter to q/v layout (B,H,L,DK): row=(b,l), col=(h,d)
// ---------------------------------------------------------------------------
__global__ __launch_bounds__(128)
void gemm_wmma_kernel(const _Float16* __restrict__ A,
                      const _Float16* __restrict__ Wt,
                      const float* __restrict__ bias,
                      float* __restrict__ out,
                      int M, int N, int K, int mode) {
    __shared__ __align__(32) _Float16 As [64][48];   // [m][k], 96B rows
    __shared__ __align__(32) _Float16 BsT[64][48];   // [n][k], K-contiguous
    const int t    = threadIdx.x;
    const int lane = t & 31;
    const int w    = t >> 5;
    const int rowBase = blockIdx.y * 64;
    const int colBase = blockIdx.x * 64;
    const u32 asOff   = (u32)(uintptr_t)&As[0][0];   // LDS byte offset

    v8f acc[4] = {};

    for (int k0 = 0; k0 < K; k0 += 32) {
        // --- A tile 64x32 via TDM (one DMA, wave 0 only)
        if (w == 0) {
            tdm_load_tile_f16(A + (size_t)rowBase * K + k0, asOff,
                              /*tile_w=*/32, /*tile_h=*/64,
                              /*tensor_w=*/(u32)K, /*tensor_h=*/(u32)M,
                              /*row_stride=*/(u32)K,
                              /*pad_en=*/1u, /*pad_interval=*/3u,
                              /*pad_amount=*/3u);
        }
        // --- W tile 32x64, stored transposed (K-contiguous per n)
        for (int i = t; i < 32 * 64; i += 128) {
            int r = i >> 6, c = i & 63;              // r = k, c = n
            BsT[c][r] = Wt[(size_t)(k0 + r) * N + (colBase + c)];
        }
        // prefetch next W tile (lowered to global_prefetch_b8)
        if (k0 + 32 < K)
            __builtin_prefetch(&Wt[(size_t)(k0 + 32) * N + colBase], 0, 1);

        if (w == 0) __builtin_amdgcn_s_wait_tensorcnt(0);
        __syncthreads();

        v16h a = load_a_frag(&As[0][0], 48, 16 * w, 0);
        #pragma unroll
        for (int nt = 0; nt < 4; ++nt) {
            v16h b = load_b_frag(&BsT[0][0], 48, nt * 16, 0);
            acc[nt] = wmma_f16(a, b, acc[nt]);
        }
        __syncthreads();
    }

    #pragma unroll
    for (int nt = 0; nt < 4; ++nt) {
        #pragma unroll
        for (int r = 0; r < 8; ++r) {
            const int m = rowBase + 16 * w + r + 8 * (lane >> 4);
            const int n = colBase + nt * 16 + (lane & 15);
            const float val = acc[nt][r] + bias[n];
            if (mode == 0) {
                out[(size_t)m * N + n] = val;
            } else {
                const int b = m >> 12, l = m & (L_ - 1);
                const int h = n >> 6,  d = n & (DK_ - 1);
                out[((((size_t)b * H_ + h) * L_) + l) * DK_ + d] = val;
            }
        }
    }
}

// ---------------------------------------------------------------------------
// Kernel 4: bucket assignment.  buckets[b,h,r,l] = argmax_n [rot, -rot]
// ---------------------------------------------------------------------------
__global__ void bucket_kernel(const float* __restrict__ q,
                              const float* __restrict__ rot,
                              int* __restrict__ buckets) {
    const int idx = blockIdx.x * blockDim.x + threadIdx.x;  // B*H*R*L
    if (idx >= B_ * H_ * R_ * L_) return;
    const int l  = idx & (L_ - 1);
    const int r  = (idx >> 12) & (R_ - 1);
    const int bh = idx >> 14;
    const float* qv = q + ((size_t)bh * L_ + l) * DK_;

    float rv[32];
    #pragma unroll
    for (int n = 0; n < 32; ++n) rv[n] = 0.f;
    for (int d = 0; d < DK_; ++d) {
        const float qd = qv[d];
        const float* rp = rot + ((size_t)r * DK_ + d) * 32;
        #pragma unroll
        for (int n = 0; n < 32; ++n) rv[n] += qd * rp[n];
    }
    float best = rv[0]; int bi = 0;
    #pragma unroll
    for (int n = 1; n < 32; ++n) if (rv[n]  > best) { best = rv[n];  bi = n; }
    #pragma unroll
    for (int n = 0; n < 32; ++n) if (-rv[n] > best) { best = -rv[n]; bi = n + 32; }
    buckets[idx] = bi;
}

// ---------------------------------------------------------------------------
// Kernel 5: stable counting sort per (b,h,r) over L=4096 items, keys in [0,64).
// ---------------------------------------------------------------------------
__global__ __launch_bounds__(128)
void lsh_sort_kernel(const int* __restrict__ buckets, int* __restrict__ ticker) {
    const int bhr = blockIdx.x;                  // B*H*R groups
    const int t   = threadIdx.x;
    const int* bk = buckets + (size_t)bhr * L_;
    int* tk       = ticker  + (size_t)bhr * L_;

    __shared__ int hist[128 * NB_];              // 32 KB
    __shared__ int base[NB_];

    for (int v = 0; v < NB_; ++v) hist[t * NB_ + v] = 0;
    __syncthreads();

    const int start = t * 32;
    for (int i = 0; i < 32; ++i) hist[t * NB_ + bk[start + i]]++;
    __syncthreads();

    if (t < NB_) {                               // per-bucket column scan
        int run = 0;
        for (int tt = 0; tt < 128; ++tt) {
            int c = hist[tt * NB_ + t];
            hist[tt * NB_ + t] = run;
            run += c;
        }
        base[t] = run;                           // bucket total
    }
    __syncthreads();
    if (t == 0) {                                // exclusive scan of totals
        int run = 0;
        for (int v = 0; v < NB_; ++v) { int c = base[v]; base[v] = run; run += c; }
    }
    __syncthreads();

    for (int i = 0; i < 32; ++i) {               // stable scatter
        const int v = bk[start + i];
        const int rank = base[v] + hist[t * NB_ + v]++;
        tk[rank] = start + i;
    }
}

// ---------------------------------------------------------------------------
// Kernel 6: chunked LSH attention. One block (128 threads, 4 waves) per
// (b,h,r,chunk). Gathers via ticker, scores+output via WMMA, scatters the
// result and LSE directly to unsorted positions (so no `undo` array).
// LDS budget ~58 KB: score buffer aliases the dead cq/ck region.
// ---------------------------------------------------------------------------
#define SENT_BIG  (-30000.0f)   // finite f16 stand-in for -1e9
#define SENT_SELF (-20000.0f)   // finite f16 stand-in for -1e5

__global__ __launch_bounds__(128)
void lsh_attn_kernel(const float* __restrict__ q,
                     const float* __restrict__ v,
                     const int* __restrict__ buckets,
                     const int* __restrict__ ticker,
                     const unsigned char* __restrict__ mask,
                     float* __restrict__ outR,
                     float* __restrict__ lseR) {
    const int bid = blockIdx.x;                  // ((b*H+h)*R + r)*NCH + n
    const int n   = bid & (NCH_ - 1);
    const int bhr = bid >> 6;
    const int bh  = bhr >> 2;
    const int b   = bhr >> 6;                    // / (H_*R_)
    const size_t tOff = (size_t)bhr * L_;

    const int t    = threadIdx.x;
    const int lane = t & 31;
    const int w    = t >> 5;

    __shared__ __align__(32) _Float16 regionA[12 * 1024];  // cq|ck, then sc
    __shared__ __align__(32) _Float16 cvT[DK_][128];       // [d][j], K-contig
    __shared__ __align__(32) _Float16 pr [C_][128];        // probabilities
    __shared__ int qb[C_], qp[C_], kb[128], kp[128], km[128];

    _Float16 (*cq)[DK_] = (_Float16(*)[DK_])(regionA);
    _Float16 (*ck)[DK_] = (_Float16(*)[DK_])(regionA + C_ * DK_);
    _Float16 (*sc)[128] = (_Float16(*)[128])(regionA);

    // ---- gather: 128 key rows (prev chunk ++ this chunk), queries = rows 64..127
    {
        const int j        = t;
        const int srcChunk = (j < C_) ? ((n + NCH_ - 1) & (NCH_ - 1)) : n;
        const int srow     = (j < C_) ? j : (j - C_);
        const int s        = srcChunk * C_ + srow;
        const int pos      = ticker[tOff + s];
        const float* qv = q + ((size_t)bh * L_ + pos) * DK_;
        const float* vv = v + ((size_t)bh * L_ + pos) * DK_;

        float nrm = 0.f;
        for (int d = 0; d < DK_; ++d) { float x = qv[d]; nrm += x * x; }
        const float inv = 1.0f / (sqrtf(nrm) + 1e-9f);
        for (int d = 0; d < DK_; ++d) {
            const float x = qv[d];
            ck[j][d]  = (_Float16)(x * inv);     // normalized key
            cvT[d][j] = (_Float16)vv[d];
        }
        kb[j] = buckets[tOff + pos];
        kp[j] = pos;
        km[j] = mask[(size_t)b * L_ + pos] ? 1 : 0;
        if (j >= C_) {
            const int i = j - C_;
            for (int d = 0; d < DK_; ++d) cq[i][d] = (_Float16)qv[d];
            qb[i] = kb[j];
            qp[i] = pos;
        }
    }
    __syncthreads();

    // ---- scores: S[64][128] = cq(64x64) * ck^T, K=64 in 2 steps of 32
    v8f accS[8] = {};
    #pragma unroll
    for (int kt = 0; kt < 2; ++kt) {
        v16h a = load_a_frag(&cq[0][0], DK_, 16 * w, kt * 32);
        #pragma unroll
        for (int nt = 0; nt < 8; ++nt) {
            v16h bf = load_b_frag(&ck[0][0], DK_, nt * 16, kt * 32);
            accS[nt] = wmma_f16(a, bf, accS[nt]);
        }
    }
    __syncthreads();                             // before writing aliased sc
    #pragma unroll
    for (int nt = 0; nt < 8; ++nt) {
        #pragma unroll
        for (int r = 0; r < 8; ++r) {
            const int m = 16 * w + r + 8 * (lane >> 4);
            const int j = nt * 16 + (lane & 15);
            sc[m][j] = (_Float16)(accS[nt][r] * 0.125f);   // / sqrt(dk)
        }
    }
    __syncthreads();

    // ---- masked softmax per query row (threads 0..63)
    if (t < C_) {
        const int i   = t;
        const int qbi = qb[i];
        const int qpi = qp[i];
        float mx = -1e30f;
        for (int j = 0; j < 128; ++j) {
            float s = (float)sc[i][j];
            const bool valid = (qbi == kb[j]) && km[j] && (kp[j] <= qpi);
            if (!valid)        s = SENT_BIG;
            if (kp[j] == qpi)  s = SENT_SELF;
            sc[i][j] = (_Float16)s;
            mx = fmaxf(mx, s);
        }
        float sum = 0.f;
        for (int j = 0; j < 128; ++j) sum += __expf((float)sc[i][j] - mx);
        const float lse = mx + __logf(sum);
        for (int j = 0; j < 128; ++j)
            pr[i][j] = (_Float16)__expf((float)sc[i][j] - lse);
        lseR[tOff + qpi] = lse;                  // scatter to unsorted position
    }
    __syncthreads();

    // ---- out[64][64] = P(64x128) * cv(128x64), K=128 in 4 steps of 32
    v8f accO[4] = {};
    #pragma unroll
    for (int kt = 0; kt < 4; ++kt) {
        v16h a = load_a_frag(&pr[0][0], 128, 16 * w, kt * 32);
        #pragma unroll
        for (int nt = 0; nt < 4; ++nt) {
            v16h bf = load_b_frag(&cvT[0][0], 128, nt * 16, kt * 32);
            accO[nt] = wmma_f16(a, bf, accO[nt]);
        }
    }
    #pragma unroll
    for (int nt = 0; nt < 4; ++nt) {
        #pragma unroll
        for (int r = 0; r < 8; ++r) {
            const int i = 16 * w + r + 8 * (lane >> 4);
            const int d = nt * 16 + (lane & 15);
            outR[(tOff + qp[i]) * DK_ + d] = accO[nt][r];
        }
    }
}

// ---------------------------------------------------------------------------
// Kernel 7: combine rounds with softmax over per-round LSE; emit f16 (B,L,DM)
// ---------------------------------------------------------------------------
__global__ void combine_kernel(const float* __restrict__ outR,
                               const float* __restrict__ lseR,
                               _Float16* __restrict__ attn) {
    const int idx = blockIdx.x * blockDim.x + threadIdx.x;   // B*H*L
    if (idx >= B_ * H_ * L_) return;
    const int l  = idx & (L_ - 1);
    const int bh = idx >> 12;
    const int b  = bh >> 4;
    const int h  = bh & (H_ - 1);

    float ls[R_], mx = -1e30f;
    #pragma unroll
    for (int r = 0; r < R_; ++r) {
        ls[r] = lseR[((size_t)bh * R_ + r) * L_ + l];
        mx = fmaxf(mx, ls[r]);
    }
    float sum = 0.f;
    #pragma unroll
    for (int r = 0; r < R_; ++r) { ls[r] = __expf(ls[r] - mx); sum += ls[r]; }
    const float inv = 1.0f / sum;

    _Float16* dst = attn + ((size_t)b * L_ + l) * DM_ + h * DK_;
    for (int d = 0; d < DK_; ++d) {
        float o = 0.f;
        #pragma unroll
        for (int r = 0; r < R_; ++r)
            o += ls[r] * inv *
                 outR[(((size_t)bh * R_ + r) * L_ + l) * DK_ + d];
        dst[d] = (_Float16)o;
    }
}

// ---------------------------------------------------------------------------
// Host-side launcher
// ---------------------------------------------------------------------------
extern "C" void kernel_launch(void* const* d_in, const int* in_sizes, int n_in,
                              void* d_out, int out_size, void* d_ws, size_t ws_size,
                              hipStream_t stream) {
    const float* x     = (const float*)d_in[0];
    const float* Wq    = (const float*)d_in[1];
    const float* bq    = (const float*)d_in[2];
    const float* Wv    = (const float*)d_in[3];
    const float* bv    = (const float*)d_in[4];
    const float* Wo    = (const float*)d_in[5];
    const float* bo    = (const float*)d_in[6];
    const float* gamma = (const float*)d_in[7];
    const float* beta  = (const float*)d_in[8];
    const float* rots  = (const float*)d_in[9];
    const unsigned char* mask = (const unsigned char*)d_in[10];
    float* out = (float*)d_out;

    // ---- workspace layout (256B aligned)
    char* ws = (char*)d_ws;
    size_t off = 0;
    auto alloc = [&](size_t bytes) {
        size_t o = off;
        off += (bytes + 255) & ~(size_t)255;
        return o;
    };
    const size_t o_normH  = alloc((size_t)B_ * L_ * DM_ * 2);         // 16 MB f16
    const size_t o_WqH    = alloc((size_t)DM_ * DM_ * 2);             //  2 MB f16
    const size_t o_WvH    = alloc((size_t)DM_ * DM_ * 2);
    const size_t o_WoH    = alloc((size_t)DM_ * DM_ * 2);
    const size_t o_q      = alloc((size_t)B_ * H_ * L_ * DK_ * 4);    // 32 MB f32
    const size_t o_v      = alloc((size_t)B_ * H_ * L_ * DK_ * 4);
    const size_t o_bkt    = alloc((size_t)B_ * H_ * R_ * L_ * 4);     //  2 MB
    const size_t o_tick   = alloc((size_t)B_ * H_ * R_ * L_ * 4);
    const size_t o_outR   = alloc((size_t)B_ * H_ * R_ * L_ * DK_ * 4); // 128 MB
    const size_t o_lseR   = alloc((size_t)B_ * H_ * R_ * L_ * 4);
    const size_t o_attnH  = alloc((size_t)B_ * L_ * DM_ * 2);         // 16 MB f16
    (void)ws_size; (void)in_sizes; (void)n_in; (void)out_size;

    _Float16* normH = (_Float16*)(ws + o_normH);
    _Float16* WqH   = (_Float16*)(ws + o_WqH);
    _Float16* WvH   = (_Float16*)(ws + o_WvH);
    _Float16* WoH   = (_Float16*)(ws + o_WoH);
    float*    qf    = (float*)(ws + o_q);
    float*    vf    = (float*)(ws + o_v);
    int*      bkt   = (int*)(ws + o_bkt);
    int*      tick  = (int*)(ws + o_tick);
    float*    outR  = (float*)(ws + o_outR);
    float*    lseR  = (float*)(ws + o_lseR);
    _Float16* attnH = (_Float16*)(ws + o_attnH);

    // 1) weight conversion
    {
        const int n = DM_ * DM_;
        const int g = (n + 255) / 256;
        cvt_f32_f16_kernel<<<g, 256, 0, stream>>>(Wq, WqH, n);
        cvt_f32_f16_kernel<<<g, 256, 0, stream>>>(Wv, WvH, n);
        cvt_f32_f16_kernel<<<g, 256, 0, stream>>>(Wo, WoH, n);
    }

    // 2) layernorm -> f16
    layernorm_kernel<<<B_ * L_, 256, 0, stream>>>(x, gamma, beta, normH);

    // 3,4) Q and V projections -> (B,H,L,DK) f32
    {
        dim3 grid(DM_ / 64, (B_ * L_) / 64);
        gemm_wmma_kernel<<<grid, 128, 0, stream>>>(normH, WqH, bq, qf,
                                                   B_ * L_, DM_, DM_, 1);
        gemm_wmma_kernel<<<grid, 128, 0, stream>>>(normH, WvH, bv, vf,
                                                   B_ * L_, DM_, DM_, 1);
    }

    // 5) bucket assignment
    {
        const int n = B_ * H_ * R_ * L_;
        bucket_kernel<<<(n + 255) / 256, 256, 0, stream>>>(qf, rots, bkt);
    }

    // 6) stable counting sort per (b,h,r)
    lsh_sort_kernel<<<B_ * H_ * R_, 128, 0, stream>>>(bkt, tick);

    // 7) chunked attention: one block per (b,h,r,chunk)
    lsh_attn_kernel<<<B_ * H_ * R_ * NCH_, 128, 0, stream>>>(
        qf, vf, bkt, tick, mask, outR, lseR);

    // 8) combine rounds -> f16 activations (B,L,DM)
    {
        const int n = B_ * H_ * L_;
        combine_kernel<<<(n + 255) / 256, 256, 0, stream>>>(outR, lseR, attnH);
    }

    // 9) output projection -> f32 d_out
    {
        dim3 grid(DM_ / 64, (B_ * L_) / 64);
        gemm_wmma_kernel<<<grid, 128, 0, stream>>>(attnH, WoH, bo, out,
                                                   B_ * L_, DM_, DM_, 0);
    }
}
